// CapsNet_89610197664525
// MI455X (gfx1250) — matrix operand order repaired and checked by reading, not compiled, not run on hardware
//
#include <hip/hip_runtime.h>

typedef __attribute__((ext_vector_type(16))) __bf16 v16bf;
typedef __attribute__((ext_vector_type(8)))  float  v8f;

#define ZBF __builtin_bit_cast(__bf16, (unsigned short)0)

__device__ inline __bf16 f2bf(float f) {
  union { float f; unsigned u; } v; v.f = f;
  unsigned x = v.u;
  unsigned r = (x + 0x7FFFu + ((x >> 16) & 1u)) >> 16;
  return __builtin_bit_cast(__bf16, (unsigned short)r);
}

// ---- WMMA fragment helpers (wave32, ISA 7.12.2 layouts) -------------------
constexpr int TLDA = 40;  // LDS leading dim for A (128 x 32 tile, padded)
constexpr int TLDB = 40;  // LDS leading dim for B' (64 x 32 tile, n-major)

__device__ inline v16bf frag_a(const __bf16* At) {
  // A 16x32 bf16. Per lane (m=lane&15, h=lane>>4): two contiguous 8-elem runs
  // k = h*8 + [0..7] and k = 16 + h*8 + [0..7]  -> 2x ds_load_b128
  int lane = threadIdx.x & 31;
  int m = lane & 15, half = lane >> 4;
  const __bf16* p = At + m * TLDA;
  v16bf a;
#pragma unroll
  for (int e = 0; e < 8; ++e) a[e] = p[half * 8 + e];
#pragma unroll
  for (int e = 0; e < 8; ++e) a[8 + e] = p[16 + half * 8 + e];
  return a;
}

__device__ inline v16bf frag_b(const __bf16* Bt, int ncol) {
  // B' stored n-major in LDS: Bt[n*TLDB + k]. Per lane (n=lane&15, h=lane>>4):
  // k = h*8 + [0..7] and k = 16 + h*8 + [0..7] -> 2x ds_load_b128
  int lane = threadIdx.x & 31;
  int n = lane & 15, half = lane >> 4;
  const __bf16* p = Bt + (ncol + n) * TLDB;
  v16bf b;
#pragma unroll
  for (int e = 0; e < 8; ++e) b[e] = p[half * 8 + e];
#pragma unroll
  for (int e = 0; e < 8; ++e) b[8 + e] = p[16 + half * 8 + e];
  return b;
}

__device__ inline v8f wmma_bf16(v16bf a, v16bf b, v8f c) {
  return __builtin_amdgcn_wmma_f32_16x16x32_bf16(false, a, false, b, (short)0, c,
                                                 false, false);
}

// ---- generic bf16 GEMM: C(f32) = act(A(MxK) * Bsw + bias) -----------------
// Bsw layout: [K/32][N][32] (32 contiguous k per (k-block, col)).
// block = 256 thr (8 waves), block tile 128x64, wave tile 32x32, K-step 32
template <int ACT>  // 0 none, 1 relu, 2 sigmoid
__global__ __launch_bounds__(256) void gemm_bf16_kernel(
    const __bf16* __restrict__ A, const __bf16* __restrict__ Bsw,
    const float* __restrict__ bias, float* __restrict__ C, int N, int K,
    int ldc, int M_real) {
  __shared__ __bf16 As[128 * TLDA];
  __shared__ __bf16 Bs[64 * TLDB];
  int tid = threadIdx.x, lane = tid & 31, wave = tid >> 5;
  int wm = wave & 3, wn = wave >> 2;
  int m0 = blockIdx.x * 128, n0 = blockIdx.y * 64;
  v8f acc[2][2] = {};
  int ar = tid >> 1, ac = (tid & 1) * 16;  // A tile 128x32: 16 elems/thread
  int br = tid >> 2, bc = (tid & 3) * 8;   // B' tile 64x32: 8 elems/thread
  for (int kc = 0; kc < K; kc += 32) {
    int kb = kc >> 5;
#pragma unroll
    for (int e = 0; e < 16; ++e)
      As[ar * TLDA + ac + e] = A[(m0 + ar) * K + kc + ac + e];
#pragma unroll
    for (int e = 0; e < 8; ++e)
      Bs[br * TLDB + bc + e] = Bsw[(kb * N + n0 + br) * 32 + bc + e];
    if (kc + 32 < K) {  // gfx1250 global_prefetch of next K slice
      __builtin_prefetch(&A[(m0 + ar) * K + kc + 32 + ac], 0, 0);
      __builtin_prefetch(&Bsw[((kb + 1) * N + n0 + br) * 32 + bc], 0, 0);
    }
    __syncthreads();
    v16bf a0 = frag_a(&As[(wm * 32) * TLDA]);
    v16bf a1 = frag_a(&As[(wm * 32 + 16) * TLDA]);
    v16bf b0 = frag_b(Bs, wn * 32);
    v16bf b1 = frag_b(Bs, wn * 32 + 16);
    acc[0][0] = wmma_bf16(a0, b0, acc[0][0]);
    acc[0][1] = wmma_bf16(a0, b1, acc[0][1]);
    acc[1][0] = wmma_bf16(a1, b0, acc[1][0]);
    acc[1][1] = wmma_bf16(a1, b1, acc[1][1]);
    __syncthreads();
  }
  int half = lane >> 4, nl = lane & 15;
#pragma unroll
  for (int mt = 0; mt < 2; ++mt)
#pragma unroll
    for (int i = 0; i < 8; ++i) {
      int m = m0 + wm * 32 + mt * 16 + i + 8 * half;
      if (m < M_real) {
#pragma unroll
        for (int nt = 0; nt < 2; ++nt) {
          int n = n0 + wn * 32 + nt * 16 + nl;
          float v = acc[mt][nt][i] + bias[n];
          if (ACT == 1) v = v > 0.f ? v : 0.f;
          else if (ACT == 2) v = 1.0f / (1.0f + __expf(-v));
          C[m * ldc + n] = v;
        }
      }
    }
}

// ---- implicit-im2col conv GEMM (MODE 0 = conv1, MODE 1 = primary caps) ----
template <int MODE>
__global__ __launch_bounds__(256) void conv_gemm_kernel(
    const __bf16* __restrict__ IN, const __bf16* __restrict__ Bsw,
    const float* __restrict__ bias, __bf16* __restrict__ OutBf,
    float* __restrict__ OutF) {
  constexpr int OHW = (MODE == 0) ? 576 : 64;
  constexpr int OW = (MODE == 0) ? 24 : 8;
  constexpr int CIN = (MODE == 0) ? 1 : 256;
  constexpr int IHW = (MODE == 0) ? 1024 : 576;
  constexpr int K = (MODE == 0) ? 96 : 20736;  // conv1 K=81 padded to 96
  constexpr int N = 256;
  __shared__ __bf16 As[128 * TLDA];
  __shared__ __bf16 Bs[64 * TLDB];
  int tid = threadIdx.x, lane = tid & 31, wave = tid >> 5;
  int wm = wave & 3, wn = wave >> 2;
  int m0 = blockIdx.x * 128, n0 = blockIdx.y * 64;
  v8f acc[2][2] = {};
  int ar = tid >> 1, ac = (tid & 1) * 16;
  int br = tid >> 2, bc = (tid & 3) * 8;
  int ma = m0 + ar;
  int ab = ma / OHW, arem = ma % OHW, aoh = arem / OW, aow = arem % OW;
  const __bf16* inb = IN + ab * (CIN * IHW);
  for (int kc = 0; kc < K; kc += 32) {
    int kb = kc >> 5;
#pragma unroll
    for (int e = 0; e < 16; ++e) {
      int k = kc + ac + e;
      __bf16 v = ZBF;
      if (MODE == 0) {
        if (k < 81) {
          int kh = k / 9, kw = k % 9;
          v = inb[(aoh + kh) * 32 + aow + kw];
        }
      } else {
        int ic = k / 81, t2 = k % 81, kh = t2 / 9, kw = t2 % 9;
        v = inb[ic * 576 + (2 * aoh + kh) * 24 + (2 * aow + kw)];
      }
      As[ar * TLDA + ac + e] = v;
    }
#pragma unroll
    for (int e = 0; e < 8; ++e)
      Bs[br * TLDB + bc + e] = Bsw[(kb * N + n0 + br) * 32 + bc + e];
    __syncthreads();
    v16bf a0 = frag_a(&As[(wm * 32) * TLDA]);
    v16bf a1 = frag_a(&As[(wm * 32 + 16) * TLDA]);
    v16bf b0 = frag_b(Bs, wn * 32);
    v16bf b1 = frag_b(Bs, wn * 32 + 16);
    acc[0][0] = wmma_bf16(a0, b0, acc[0][0]);
    acc[0][1] = wmma_bf16(a0, b1, acc[0][1]);
    acc[1][0] = wmma_bf16(a1, b0, acc[1][0]);
    acc[1][1] = wmma_bf16(a1, b1, acc[1][1]);
    __syncthreads();
  }
  int half = lane >> 4, nl = lane & 15;
#pragma unroll
  for (int mt = 0; mt < 2; ++mt)
#pragma unroll
    for (int i = 0; i < 8; ++i) {
      int m = m0 + wm * 32 + mt * 16 + i + 8 * half;
      int b = m / OHW, rem = m % OHW, oh = rem / OW, ow = rem % OW;
#pragma unroll
      for (int nt = 0; nt < 2; ++nt) {
        int n = n0 + wn * 32 + nt * 16 + nl;
        float v = acc[mt][nt][i] + bias[n];
        if (MODE == 0) {
          v = v > 0.f ? v : 0.f;
          OutBf[((b * 256 + n) * 24 + oh) * 24 + ow] = f2bf(v);
        } else {
          OutF[((b * 256 + n) * 8 + oh) * 8 + ow] = v;  // == u[b][n*8+oh][ow]
        }
      }
    }
}

// ---- small elementwise kernels --------------------------------------------
__global__ void cvt_f32_bf16_kernel(const float* __restrict__ s,
                                    __bf16* __restrict__ d, int n) {
  int i = blockIdx.x * blockDim.x + threadIdx.x;
  if (i < n) d[i] = f2bf(s[i]);
}

// W (N x K) f32 row-major -> Bsw (Kpad/32 x N x 32) bf16, zero-padded K tail
__global__ void wt_kernel(const float* __restrict__ W, __bf16* __restrict__ out,
                          int N, int K, int Kpad) {
  int i = blockIdx.x * blockDim.x + threadIdx.x;
  if (i < Kpad * N) {
    int kb = i / (N * 32), rem = i % (N * 32), n = rem / 32, kk = rem % 32;
    int k = kb * 32 + kk;
    out[i] = (k < K) ? f2bf(W[n * K + k]) : ZBF;
  }
}

__global__ void zero_kernel(float* p, int n) {
  int i = blockIdx.x * blockDim.x + threadIdx.x;
  if (i < n) p[i] = 0.f;
}

// ---- primary-caps squash: per batch, bitonic sort 2048, aliased piecewise -
#define PT1 (-13.46416092f)
#define PA1 (0.000242759f)
#define PB1 (0.024488359f)
#define PA2 (0.002769205f)
#define PB2 (0.06089699f)
#define PT3 (13.23405266f)
#define PA3 (-0.002828244f)
#define PB3 (0.061313814f)
#define PA4 (-0.000219038f)
#define PB4 (0.023874787f)

__global__ __launch_bounds__(256) void squash_prim_kernel(
    const float* __restrict__ u, float* __restrict__ u_sq) {
  __shared__ float key[2048];
  __shared__ int kidx[2048];
  __shared__ float magl[2048];
  __shared__ int red[256];
  int b = blockIdx.x, tid = threadIdx.x;
  for (int t = tid; t < 2048; t += 256) {
    key[t] = u[(b * 2048 + t) * 8];
    kidx[t] = t;
  }
  __syncthreads();
  for (int ksz = 2; ksz <= 2048; ksz <<= 1)
    for (int j = ksz >> 1; j > 0; j >>= 1) {
      for (int t = tid; t < 2048; t += 256) {
        int ixj = t ^ j;
        if (ixj > t) {
          bool up = ((t & ksz) == 0);
          float a = key[t], c = key[ixj];
          if ((a > c) == up) {
            key[t] = c; key[ixj] = a;
            int ta = kidx[t]; kidx[t] = kidx[ixj]; kidx[ixj] = ta;
          }
        }
      }
      __syncthreads();
    }
  auto count_lt = [&](float thr) -> int {
    int c = 0;
    for (int t = tid; t < 2048; t += 256) c += (key[t] < thr) ? 1 : 0;
    red[tid] = c;
    __syncthreads();
    for (int s = 128; s > 0; s >>= 1) {
      if (tid < s) red[tid] += red[tid + s];
      __syncthreads();
    }
    int tot = red[0];
    __syncthreads();
    return tot;
  };
  // faithful to the aliased reference: counts on partially-modified array
  int i1 = count_lt(PT1);
  for (int t = tid; t < 2048; t += 256)
    if (t < i1 - 1) key[t] = PA1 * key[t] + PB1;
  __syncthreads();
  int i2 = count_lt(0.0f);
  for (int t = tid; t < 2048; t += 256)
    if (t >= i1 && t < i2 - 1) key[t] = PA2 * key[t] + PB2;
  __syncthreads();
  int i3 = count_lt(PT3);
  for (int t = tid; t < 2048; t += 256)
    if (t >= i2 && t < i3 - 1) key[t] = PA3 * key[t] + PB3;
  __syncthreads();
  for (int t = tid; t < 2048; t += 256)
    if (t >= i3 && t < 2047) key[t] = PA4 * key[t] + PB4;
  __syncthreads();
  for (int t = tid; t < 2048; t += 256) magl[kidx[t]] = key[t];  // unsort
  __syncthreads();
  for (int e = tid; e < 16384; e += 256)
    u_sq[b * 16384 + e] = magl[e >> 3] * u[b * 16384 + e];
}

// ---- u_hat[b,r,c,o] = sum_i W_dc[r,c,o,i] * u_sq[b,r,i] -------------------
__global__ __launch_bounds__(160) void u_hat_kernel(
    const float* __restrict__ W_dc, const float* __restrict__ u_sq,
    float* __restrict__ u_hat) {
  int r = blockIdx.x, tid = threadIdx.x;  // tid = c*16+o
  __shared__ float us[8];
  float w[8];
#pragma unroll
  for (int i = 0; i < 8; ++i) w[i] = W_dc[(r * 160 + tid) * 8 + i];
  for (int b = 0; b < 100; ++b) {
    if (tid < 8) us[tid] = u_sq[(b * 2048 + r) * 8 + tid];
    __syncthreads();
    float s = 0.f;
#pragma unroll
    for (int i = 0; i < 8; ++i) s += w[i] * us[i];
    u_hat[(b * 2048 + r) * 160 + tid] = s;
    __syncthreads();
  }
}

// ---- softmax over 2048 routes per class -----------------------------------
__global__ __launch_bounds__(256) void softmax_routes_kernel(
    const float* __restrict__ b_ij, float* __restrict__ c_ij) {
  int c = blockIdx.x, tid = threadIdx.x;
  __shared__ float red[256];
  float m = -1e30f;
  for (int r = tid; r < 2048; r += 256) m = fmaxf(m, b_ij[r * 10 + c]);
  red[tid] = m;
  __syncthreads();
  for (int s = 128; s > 0; s >>= 1) {
    if (tid < s) red[tid] = fmaxf(red[tid], red[tid + s]);
    __syncthreads();
  }
  m = red[0];
  __syncthreads();
  float sum = 0.f;
  for (int r = tid; r < 2048; r += 256) sum += __expf(b_ij[r * 10 + c] - m);
  red[tid] = sum;
  __syncthreads();
  for (int s = 128; s > 0; s >>= 1) {
    if (tid < s) red[tid] += red[tid + s];
    __syncthreads();
  }
  float inv = 1.f / red[0];
  __syncthreads();
  for (int r = tid; r < 2048; r += 256)
    c_ij[r * 10 + c] = __expf(b_ij[r * 10 + c] - m) * inv;
}

// ---- s_j[b,c,:] = sum_r c_ij[r,c] * u_hat[b,r,c,:] ------------------------
__global__ __launch_bounds__(256) void s_j_kernel(
    const float* __restrict__ c_ij, const float* __restrict__ u_hat,
    float* __restrict__ s_j) {
  int b = blockIdx.x, c = blockIdx.y, tid = threadIdx.x;
  __shared__ float red[256 * 16];
  float acc[16];
#pragma unroll
  for (int o = 0; o < 16; ++o) acc[o] = 0.f;
  for (int r = tid; r < 2048; r += 256) {
    float cv = c_ij[r * 10 + c];
    const float* uh = &u_hat[((b * 2048 + r) * 10 + c) * 16];
#pragma unroll
    for (int o = 0; o < 16; ++o) acc[o] += cv * uh[o];
  }
#pragma unroll
  for (int o = 0; o < 16; ++o) red[tid * 16 + o] = acc[o];
  __syncthreads();
  for (int s = 128; s > 0; s >>= 1) {
    if (tid < s) {
#pragma unroll
      for (int o = 0; o < 16; ++o) red[tid * 16 + o] += red[(tid + s) * 16 + o];
    }
    __syncthreads();
  }
  if (tid < 16) s_j[(b * 10 + c) * 16 + tid] = red[tid];
}

// ---- digit squash + v_j (optionally writes v_j slice of d_out) ------------
#define DT1 (-0.075410217f)
#define DA1 (-0.074520095f)
#define DB1 (0.349297946f)
#define DA2 (-0.534473989f)
#define DB2 (0.27196494f)
#define DT3 (0.062207676f)
#define DA3 (0.637642944f)
#define DB3 (0.295330779f)
#define DA4 (0.169344703f)
#define DB4 (0.353784456f)

__global__ __launch_bounds__(160) void digit_squash_kernel(
    const float* __restrict__ s_j, float* __restrict__ v_j,
    float* __restrict__ out, int write_out) {
  int b = blockIdx.x, tid = threadIdx.x;
  __shared__ float fl[10];
  if (tid == 0) {
    float s[10]; int id[10];
    for (int c = 0; c < 10; ++c) { s[c] = s_j[(b * 10 + c) * 16]; id[c] = c; }
    for (int i = 1; i < 10; ++i) {  // insertion sort ascending
      float kv = s[i]; int ki = id[i]; int j = i - 1;
      while (j >= 0 && s[j] > kv) { s[j + 1] = s[j]; id[j + 1] = id[j]; --j; }
      s[j + 1] = kv; id[j + 1] = ki;
    }
    int i1 = 0; for (int t = 0; t < 10; ++t) i1 += (s[t] < DT1);
    for (int t = 0; t < 10; ++t) if (t < i1 - 1) s[t] = DA1 * s[t] + DB1;
    int i2 = 0; for (int t = 0; t < 10; ++t) i2 += (s[t] < 0.f);
    for (int t = 0; t < 10; ++t) if (t >= i1 && t < i2 - 1) s[t] = DA2 * s[t] + DB2;
    int i3 = 0; for (int t = 0; t < 10; ++t) i3 += (s[t] < DT3);
    for (int t = 0; t < 10; ++t) if (t >= i2 && t < i3 - 1) s[t] = DA3 * s[t] + DB3;
    for (int t = 0; t < 10; ++t) if (t >= i3 && t < 9) s[t] = DA4 * s[t] + DB4;
    for (int t = 0; t < 10; ++t) fl[id[t]] = s[t];
  }
  __syncthreads();
  int c = tid >> 4, o = tid & 15;
  float fv = fl[c];
  float sm = (o == 0) ? fv : s_j[(b * 10 + c) * 16 + o];  // aliased writeback
  float v = fv * sm;
  v_j[(b * 10 + c) * 16 + o] = v;
  if (write_out) out[b * 1184 + tid] = v;
}

// ---- agreement: b_ij[r,c] += mean_b dot16(u_hat[b,r,c,:], v_j[b,c,:]) -----
__global__ __launch_bounds__(256) void agreement_kernel(
    const float* __restrict__ u_hat, const float* __restrict__ v_j,
    float* __restrict__ b_ij) {
  int r = blockIdx.x, tid = threadIdx.x;
  __shared__ float bsum[10];
  if (tid < 10) bsum[tid] = 0.f;
  __syncthreads();
  for (int p = tid; p < 1000; p += 256) {
    int b = p / 10, c = p % 10;
    const float* uh = &u_hat[((b * 2048 + r) * 10 + c) * 16];
    const float* vv = &v_j[(b * 10 + c) * 16];
    float d = 0.f;
#pragma unroll
    for (int o = 0; o < 16; ++o) d += uh[o] * vv[o];
    atomicAdd(&bsum[c], d);
  }
  __syncthreads();
  if (tid < 10) b_ij[r * 10 + tid] += bsum[tid] * 0.01f;
}

// ---- decoder mask: norms, softmax over batch, argmax, build t (128x160) ---
__global__ __launch_bounds__(1024) void decoder_mask_kernel(
    const float* __restrict__ v_j, __bf16* __restrict__ t_bf) {
  __shared__ float cls[1000];
  __shared__ int idxl[100];
  int tid = threadIdx.x;
  if (tid < 1000) {
    const float* vv = &v_j[tid * 16];
    float s = 0.f;
#pragma unroll
    for (int o = 0; o < 16; ++o) s += vv[o] * vv[o];
    cls[tid] = sqrtf(s);
  }
  __syncthreads();
  if (tid < 10) {  // softmax over batch axis, per class
    int c = tid;
    float m = -1e30f;
    for (int b = 0; b < 100; ++b) m = fmaxf(m, cls[b * 10 + c]);
    float s = 0.f;
    for (int b = 0; b < 100; ++b) s += __expf(cls[b * 10 + c] - m);
    float inv = 1.f / s;
    for (int b = 0; b < 100; ++b) cls[b * 10 + c] = __expf(cls[b * 10 + c] - m) * inv;
  }
  __syncthreads();
  if (tid < 100) {
    int best = 0; float bv = cls[tid * 10];
    for (int c = 1; c < 10; ++c) {
      float v = cls[tid * 10 + c];
      if (v > bv) { bv = v; best = c; }
    }
    idxl[tid] = best;
  }
  __syncthreads();
  for (int e = tid; e < 128 * 160; e += 1024) {
    int b = e / 160, co = e % 160, c = co >> 4;
    float v = 0.f;
    if (b < 100 && c == idxl[b]) v = v_j[b * 160 + co];
    t_bf[e] = f2bf(v);
  }
}

// ---------------------------------------------------------------------------
extern "C" void kernel_launch(void* const* d_in, const int* in_sizes, int n_in,
                              void* d_out, int out_size, void* d_ws,
                              size_t ws_size, hipStream_t stream) {
  const float* data = (const float*)d_in[0];
  const float* conv1_w = (const float*)d_in[1];
  const float* conv1_b = (const float*)d_in[2];
  const float* prim_w = (const float*)d_in[3];
  const float* prim_b = (const float*)d_in[4];
  const float* W_dc = (const float*)d_in[5];
  const float* dec_w1 = (const float*)d_in[6];
  const float* dec_b1 = (const float*)d_in[7];
  const float* dec_w2 = (const float*)d_in[8];
  const float* dec_b2 = (const float*)d_in[9];
  const float* dec_w3 = (const float*)d_in[10];
  const float* dec_b3 = (const float*)d_in[11];
  float* out = (float*)d_out;

  char* ws = (char*)d_ws;
  size_t off = 0;
  auto alloc = [&](size_t bytes) -> void* {
    void* p = (void*)(ws + off);
    off += (bytes + 255) & ~(size_t)255;
    return p;
  };
  __bf16* data_bf = (__bf16*)alloc(102400 * 2);
  __bf16* B1 = (__bf16*)alloc(96 * 256 * 2);
  __bf16* Bp = (__bf16*)alloc((size_t)20736 * 256 * 2);
  __bf16* Bd1 = (__bf16*)alloc(160 * 512 * 2);
  __bf16* Bd2 = (__bf16*)alloc(512 * 1024 * 2);
  __bf16* Bd3 = (__bf16*)alloc(1024 * 1024 * 2);
  __bf16* x_bf = (__bf16*)alloc((size_t)14745600 * 2);
  float* u = (float*)alloc((size_t)1638400 * 4);
  float* u_sq = (float*)alloc((size_t)1638400 * 4);
  float* u_hat = (float*)alloc((size_t)32768000 * 4);
  float* b_ij = (float*)alloc(20480 * 4);
  float* c_ij = (float*)alloc(20480 * 4);
  float* s_j = (float*)alloc(16000 * 4);
  float* v_j = (float*)alloc(16000 * 4);
  __bf16* t_bf = (__bf16*)alloc(128 * 160 * 2);
  float* h1 = (float*)alloc(128 * 512 * 4);
  __bf16* h1b = (__bf16*)alloc(128 * 512 * 2);
  float* h2 = (float*)alloc(128 * 1024 * 4);
  __bf16* h2b = (__bf16*)alloc(128 * 1024 * 2);

  dim3 blk(256);
  // input + weight packing to bf16 (B matrices stored [K/32][N][32] swizzled)
  cvt_f32_bf16_kernel<<<(102400 + 255) / 256, blk, 0, stream>>>(data, data_bf, 102400);
  wt_kernel<<<(96 * 256 + 255) / 256, blk, 0, stream>>>(conv1_w, B1, 256, 81, 96);
  wt_kernel<<<(20736 * 256 + 255) / 256, blk, 0, stream>>>(prim_w, Bp, 256, 20736, 20736);
  wt_kernel<<<(160 * 512 + 255) / 256, blk, 0, stream>>>(dec_w1, Bd1, 512, 160, 160);
  wt_kernel<<<(512 * 1024 + 255) / 256, blk, 0, stream>>>(dec_w2, Bd2, 1024, 512, 512);
  wt_kernel<<<(1024 * 1024 + 255) / 256, blk, 0, stream>>>(dec_w3, Bd3, 1024, 1024, 1024);

  // conv1: M=57600 (b,oh,ow), N=256, K=81->96, relu, bf16 out
  conv_gemm_kernel<0><<<dim3(450, 4), blk, 0, stream>>>(data_bf, B1, conv1_b, x_bf, nullptr);
  // primary conv: M=6400, N=256, K=20736, f32 out == u (B,2048,8)
  conv_gemm_kernel<1><<<dim3(50, 4), blk, 0, stream>>>(x_bf, Bp, prim_b, nullptr, u);

  squash_prim_kernel<<<100, blk, 0, stream>>>(u, u_sq);
  u_hat_kernel<<<2048, 160, 0, stream>>>(W_dc, u_sq, u_hat);

  zero_kernel<<<(20480 + 255) / 256, blk, 0, stream>>>(b_ij, 20480);
  for (int it = 0; it < 3; ++it) {
    softmax_routes_kernel<<<10, blk, 0, stream>>>(b_ij, c_ij);
    s_j_kernel<<<dim3(100, 10), blk, 0, stream>>>(c_ij, u_hat, s_j);
    digit_squash_kernel<<<100, 160, 0, stream>>>(s_j, v_j, out, it == 2 ? 1 : 0);
    if (it < 2) agreement_kernel<<<2048, blk, 0, stream>>>(u_hat, v_j, b_ij);
  }

  // decoder (buffers padded to 128 rows; only rows <100 stored to d_out)
  decoder_mask_kernel<<<1, 1024, 0, stream>>>(v_j, t_bf);
  gemm_bf16_kernel<1><<<dim3(1, 8), blk, 0, stream>>>(t_bf, Bd1, dec_b1, h1, 512, 160, 512, 128);
  cvt_f32_bf16_kernel<<<(65536 + 255) / 256, blk, 0, stream>>>(h1, h1b, 65536);
  gemm_bf16_kernel<1><<<dim3(1, 16), blk, 0, stream>>>(h1b, Bd2, dec_b2, h2, 1024, 512, 1024, 128);
  cvt_f32_bf16_kernel<<<(131072 + 255) / 256, blk, 0, stream>>>(h2, h2b, 131072);
  // final layer: sigmoid, strided store into d_out (rows of 1184, offset 160)
  gemm_bf16_kernel<2><<<dim3(1, 16), blk, 0, stream>>>(h2b, Bd3, dec_b3, out + 160, 1024, 1024, 1184, 100);
}
